// BmmEnsemble_36661840839141
// MI455X (gfx1250) — matrix operand order repaired and checked by reading, not compiled, not run on hardware
//
#include <hip/hip_runtime.h>
#include <hip/hip_bf16.h>

// ---------------------------------------------------------------------------
// ANI ensemble MLP on gfx1250 (CDNA5).
//  1) bucket atoms by species (only own-species net contributes to the energy)
//  2) convert AEV f32 -> f16 once (halves HBM activation reads)
//  3) pre-swizzle f32 weights -> f16 WMMA B-fragments (per-lane contiguous)
//  4) main kernel: block = 128 atoms x ONE (species, ensemble) pair;
//     8 waves each own a 16-atom M-tile and share the SAME weight stream
//     (WGP$ hit after first miss -> 8x less L2 weight traffic).
//     Layer-1 A comes straight from global f16 AEV (row gather, register-
//     hoisted across N-tiles); h1/h2 transit per-wave LDS; f32 accumulation;
//     CELU between layers; masked atomic energy sum with 1/E folded in.
// ---------------------------------------------------------------------------

#define S_  4
#define E_  8
#define D_  384
#define H1_ 160
#define H2_ 128
#define H3_ 96
#define ALPHA_ 0.1f

typedef __attribute__((ext_vector_type(16))) _Float16 v16h;
typedef __attribute__((ext_vector_type(8)))  _Float16 v8h;
typedef __attribute__((ext_vector_type(8)))  float    v8f;

// ---------------- init: zero species counters and output scalar -------------
__global__ void ani_init_kernel(int* counts, float* out) {
    if (threadIdx.x < S_) counts[threadIdx.x] = 0;
    if (threadIdx.x == 0) out[0] = 0.0f;
}

// ---------------- bucket atoms by species ----------------------------------
__global__ void ani_bucket_kernel(const int* __restrict__ species,
                                  int* __restrict__ counts,
                                  int* __restrict__ idx, int n) {
    int i = blockIdx.x * blockDim.x + threadIdx.x;
    if (i < n) {
        int sp = species[i] & 3;
        int pos = atomicAdd(&counts[sp], 1);
        idx[sp * n + pos] = i;
    }
}

// ---------------- AEV f32 -> f16 --------------------------------------------
__global__ void ani_cvt_kernel(const float* __restrict__ src,
                               _Float16* __restrict__ dst, int n) {
    int i = blockIdx.x * blockDim.x + threadIdx.x;
    if (i < n) dst[i] = (_Float16)src[i];
}

// ---------------- weight swizzle: f32 [SE][K][N] -> f16 B fragments ---------
// Block b = ((se*NT + nt)*KT + kt). Within a block: 512 halfs, lane l owns
// halfs [l*16..l*16+15]; half j of lane l is W[kt*32 + (l>=16)*8 + (j&7) + (j>>3)*16][nt*16 + (l&15)].
__global__ void ani_swizzle_kernel(const float* __restrict__ W,
                                   _Float16* __restrict__ dst,
                                   int Kdim, int Nsrc, int NT, int KT) {
    int b = blockIdx.x;
    int tid = threadIdx.x;            // 0..511
    int kt = b % KT;  int t = b / KT;
    int nt = t % NT;  int se = t / NT;
    int lane = tid >> 4;              // 0..31
    int j = tid & 15;
    int K = kt * 32 + ((lane >> 4) << 3) + (j & 7) + ((j >> 3) << 4);      // +8 for lanes 16..31
    int N = nt * 16 + (lane & 15);
    float v = 0.0f;
    if (N < Nsrc) v = W[(size_t)se * Kdim * Nsrc + (size_t)K * Nsrc + N];
    dst[(size_t)b * 512 + tid] = (_Float16)v;
}

// ---------------- fragment loaders -----------------------------------------
// A fragment at a precomputed per-lane base (row*stride + (lane>=16)*8 halfs).
__device__ __forceinline__ v16h load_a_at(const _Float16* pBase, int kstep) {
    const _Float16* p = pBase + kstep * 32;
    v8h lo = *(const v8h*)(p);        // K = base+0..7
    v8h hi = *(const v8h*)(p + 16);   // K = base+16..23
    return __builtin_shufflevector(lo, hi, 0,1,2,3,4,5,6,7,8,9,10,11,12,13,14,15);
}

// B (32x16 f16) from pre-swizzled global weights: 32 contiguous bytes / lane.
__device__ __forceinline__ v16h load_b(const _Float16* blockBase, int lane) {
    const _Float16* p = blockBase + lane * 16;
    v8h lo = *(const v8h*)(p);
    v8h hi = *(const v8h*)(p + 8);
    return __builtin_shufflevector(lo, hi, 0,1,2,3,4,5,6,7,8,9,10,11,12,13,14,15);
}

__device__ __forceinline__ float celu_f(float x) {
    return x > 0.0f ? x : ALPHA_ * (__expf(x * (1.0f / ALPHA_)) - 1.0f);
}

// ---------------- one hidden layer: A(base ptr) x global B -> LDS out (CELU)
template <int KDIM, int NDIM>
__device__ __forceinline__ void mlp_layer(const _Float16* __restrict__ aBase,
                                          const _Float16* __restrict__ wSE,
                                          const float* __restrict__ bias,
                                          _Float16* __restrict__ outDst,
                                          int lane) {
    constexpr int KT = KDIM / 32;
    constexpr int NT = NDIM / 16;
    int col0 = lane & 15;
    int rbase = (lane >> 4) * 8;
    for (int nt = 0; nt < NT; ++nt) {
        v8f acc = {};
        const _Float16* wb = wSE + (size_t)nt * KT * 512;
        #pragma unroll
        for (int kt = 0; kt < KT; ++kt) {
            v16h a = load_a_at(aBase, kt);
            v16h b = load_b(wb + kt * 512, lane);
            acc = __builtin_amdgcn_wmma_f32_16x16x32_f16(
                false, a, false, b, (short)0, acc, false, false);
        }
        int col = nt * 16 + col0;
        float bv = bias[col];
        #pragma unroll
        for (int v = 0; v < 8; ++v) {
            float y = celu_f(acc[v] + bv);
            outDst[(rbase + v) * NDIM + col] = (_Float16)y;
        }
    }
    // Same-wave LDS write->read ordering for the next layer's A loads.
    asm volatile("s_wait_dscnt 0" ::: "memory");
}

// ---------------- main kernel: 128 atoms x one (s,e) per block --------------
__global__ __launch_bounds__(256) void ani_mlp_kernel(
    const _Float16* __restrict__ aev16,
    const int* __restrict__ idx,      // [S][N]
    const int* __restrict__ counts,   // [S]
    const _Float16* __restrict__ w1s, const _Float16* __restrict__ w2s,
    const _Float16* __restrict__ w3s, const _Float16* __restrict__ w4s,
    const float* __restrict__ b1, const float* __restrict__ b2,
    const float* __restrict__ b3, const float* __restrict__ b4,
    float* __restrict__ out, int nAtoms) {
    extern __shared__ _Float16 lds[];

    const int s   = blockIdx.z;
    const int e   = blockIdx.y;
    const int cnt = counts[s];
    const int w   = threadIdx.x >> 5;          // wave id 0..7 (wave32)
    const int lane = threadIdx.x & 31;
    const int base = blockIdx.x * 128 + w * 16;  // this wave's 16-atom tile
    if (base >= cnt) return;                     // no barriers: safe per-wave exit

    const int se = s * E_ + e;
    _Float16* hA = lds + w * (16 * H1_ + 16 * H2_);  // h1 / h3 (16 x 160)
    _Float16* hB = hA + 16 * H1_;                    // h2       (16 x 128)

    // Layer-1 A base: gather this lane's atom row from global f16 AEV.
    const int r  = lane & 15;
    const int ga = base + r;
    const int atom = idx[s * nAtoms + ((ga < cnt) ? ga : 0)];  // padded rows masked later
    const _Float16* aRow = aev16 + (size_t)atom * D_ + (lane >> 4) * 8;
    const _Float16* hARow = hA + r * H1_ + (lane >> 4) * 8;
    const _Float16* hBRow = hB + r * H2_ + (lane >> 4) * 8;
    const _Float16* h3Row = hA + r * H3_ + (lane >> 4) * 8;   // hA reused for h3

    constexpr int KT1 = D_ / 32,  NT1 = H1_ / 16;
    constexpr int KT2 = H1_ / 32, NT2 = H2_ / 16;
    constexpr int KT3 = H2_ / 32, NT3 = H3_ / 16;
    constexpr int KT4 = H3_ / 32;

    mlp_layer<D_, H1_>(aRow, w1s + (size_t)se * NT1 * KT1 * 512,
                       b1 + (size_t)se * H1_, hA, lane);
    mlp_layer<H1_, H2_>(hARow, w2s + (size_t)se * NT2 * KT2 * 512,
                        b2 + (size_t)se * H2_, hB, lane);
    mlp_layer<H2_, H3_>(hBRow, w3s + (size_t)se * NT3 * KT3 * 512,
                        b3 + (size_t)se * H3_, hA, lane);

    // Layer 4: K=96, N padded to 16 (only column 0 is real).
    v8f acc = {};
    const _Float16* w4b = w4s + (size_t)se * KT4 * 512;
    #pragma unroll
    for (int kt = 0; kt < KT4; ++kt) {
        v16h a = load_a_at(h3Row, kt);
        v16h b = load_b(w4b + kt * 512, lane);
        acc = __builtin_amdgcn_wmma_f32_16x16x32_f16(
            false, a, false, b, (short)0, acc, false, false);
    }
    if ((lane & 15) == 0) {               // column 0 holds the 16 atom energies
        float bv = b4[se];
        int rbase = (lane >> 4) * 8;
        float part = 0.0f;
        #pragma unroll
        for (int v = 0; v < 8; ++v)
            if (base + rbase + v < cnt) part += acc[v] + bv;
        atomicAdd(out, part * (1.0f / E_));  // ensemble mean folded in
    }
}

// ---------------------------------------------------------------------------
extern "C" void kernel_launch(void* const* d_in, const int* in_sizes, int n_in,
                              void* d_out, int out_size, void* d_ws, size_t ws_size,
                              hipStream_t stream) {
    const int N = in_sizes[0];                      // 40000 atoms
    const int*   species = (const int*)d_in[0];
    const float* aev     = (const float*)d_in[1];
    const float* W1 = (const float*)d_in[2];
    const float* b1 = (const float*)d_in[3];
    const float* W2 = (const float*)d_in[4];
    const float* b2 = (const float*)d_in[5];
    const float* W3 = (const float*)d_in[6];
    const float* b3 = (const float*)d_in[7];
    const float* W4 = (const float*)d_in[8];
    const float* b4 = (const float*)d_in[9];
    float* out = (float*)d_out;

    // Workspace layout
    char* ws = (char*)d_ws;
    size_t off = 0;
    int* counts = (int*)(ws + off); off += 256;
    int* idxbuf = (int*)(ws + off); off += (size_t)S_ * N * sizeof(int);
    off = (off + 255) & ~(size_t)255;
    _Float16* aev16 = (_Float16*)(ws + off); off += (size_t)N * D_ * 2;
    off = (off + 255) & ~(size_t)255;
    const int NT1 = H1_/16, KT1 = D_/32;
    const int NT2 = H2_/16, KT2 = H1_/32;
    const int NT3 = H3_/16, KT3 = H2_/32;
    const int NT4 = 1,      KT4 = H3_/32;
    _Float16* w1s = (_Float16*)(ws + off); off += (size_t)S_*E_*NT1*KT1*512*2;
    _Float16* w2s = (_Float16*)(ws + off); off += (size_t)S_*E_*NT2*KT2*512*2;
    _Float16* w3s = (_Float16*)(ws + off); off += (size_t)S_*E_*NT3*KT3*512*2;
    _Float16* w4s = (_Float16*)(ws + off); off += (size_t)S_*E_*NT4*KT4*512*2;
    (void)ws_size; (void)n_in; (void)out_size;

    ani_init_kernel<<<1, 64, 0, stream>>>(counts, out);
    ani_bucket_kernel<<<(N + 255) / 256, 256, 0, stream>>>(species, counts, idxbuf, N);
    ani_cvt_kernel<<<(N * D_ + 255) / 256, 256, 0, stream>>>(aev, aev16, N * D_);

    ani_swizzle_kernel<<<S_*E_*NT1*KT1, 512, 0, stream>>>(W1, w1s, D_,  H1_, NT1, KT1);
    ani_swizzle_kernel<<<S_*E_*NT2*KT2, 512, 0, stream>>>(W2, w2s, H1_, H2_, NT2, KT2);
    ani_swizzle_kernel<<<S_*E_*NT3*KT3, 512, 0, stream>>>(W3, w3s, H2_, H3_, NT3, KT3);
    ani_swizzle_kernel<<<S_*E_*NT4*KT4, 512, 0, stream>>>(W4, w4s, H3_, 1,   NT4, KT4);

    // LDS: per wave h1 (16x160) + h2 (16x128) in f16 -> 8 * 9216 B = 73728 B
    size_t shmem = (size_t)E_ * (16 * H1_ + 16 * H2_) * sizeof(_Float16);
    dim3 grid((N + 127) / 128, E_, S_);
    ani_mlp_kernel<<<grid, 256, shmem, stream>>>(
        aev16, idxbuf, counts, w1s, w2s, w3s, w4s, b1, b2, b3, b4, out, N);
}
